// ScaledDotProductAttention_27908697490002
// MI455X (gfx1250) — matrix operand order repaired and checked by reading, not compiled, not run on hardware
//
#include <hip/hip_runtime.h>

// CDNA5 / gfx1250 WMMA vector types
typedef __attribute__((ext_vector_type(16))) _Float16     v16h;
typedef __attribute__((ext_vector_type(8)))  float        v8f;
typedef __attribute__((ext_vector_type(4)))  unsigned int u32x4;
typedef __attribute__((ext_vector_type(8)))  unsigned int u32x8;

#define B_ 4
#define H_ 8
#define S_ 2048
#define D_ 64

// Padded LDS row stride (floats) for the score tile: 2052 % 64 == 4 -> the 16
// rows land on distinct banks for the strided A-fragment reads in phase 3.
#define SSTRIDE 2052
#define SCORE_F (16 * SSTRIDE)      // 16 x 2048 f32 scores (padded)
#define RED_F   32                  // cross-wave reduction scratch
// V staging: TDM writes 32 rows x 64 f32 with 1-DWORD pad per row -> stride 65.
#define VSTRIDE 65
#define VBUF_F  (32 * VSTRIDE)      // one f32 V chunk buffer (padded)
#define NVBUF   2                   // double-buffered TDM
#define SMEM_BYTES ((SCORE_F + RED_F + NVBUF * VBUF_F) * 4)   // 148,096 B < 320 KB/WGP

// ---------------------------------------------------------------------------
// TDM: DMA one 32x64 f32 V chunk from global memory into LDS (padded rows).
// D# layout per CDNA5 ISA.  2D tile -> groups 0+1 only.
// Inline asm (not the clang builtin) for toolchain portability.
// ---------------------------------------------------------------------------
__device__ __forceinline__ void tdm_load_v_chunk(unsigned lds_addr,
                                                 unsigned long long gaddr)
{
    u32x4 g0;
    g0[0] = 1u;                                   // count=1, user descriptor
    g0[1] = lds_addr;                             // lds_addr[31:0] (bytes)
    g0[2] = (unsigned)gaddr;                      // global_addr[31:0]
    g0[3] = (unsigned)((gaddr >> 32) & 0x01FFFFFFull) | (2u << 30); // [56:32]+type=2

    u32x8 g1;
    g1[0] = (2u << 16)        // data_size = 2 -> 4-byte elements
          | (1u << 20)        // pad_enable: pad LDS rows
          | (5u << 22);       // pad_interval = 5 -> every 64 DWORDs (one row)
                              // pad_amount  = 0 -> 1 DWORD  => LDS stride 65
    g1[1] = (64u & 0xFFFFu) << 16;     // tensor_dim0 = 64 (low16; bits 79:48)
    g1[2] = (2048u & 0xFFFFu) << 16;   // tensor_dim0 hi=0 | tensor_dim1=2048 low16
    g1[3] = (64u << 16);               // tensor_dim1 hi=0 | tile_dim0 = 64
    g1[4] = 32u;                       // tile_dim1 = 32 | tile_dim2 = 0
    g1[5] = 64u;                       // tensor_dim0_stride = 64 (low32)
    g1[6] = 0u;                        // dim0_stride hi | dim1_stride low (unused, 2D)
    g1[7] = 0u;

    asm volatile("tensor_load_to_lds %0, %1" :: "s"(g0), "s"(g1) : "memory");
}

__launch_bounds__(128, 1)
__global__ void sdpa_wmma_kernel(const float* __restrict__ Q,
                                 const float* __restrict__ K,
                                 const float* __restrict__ V,
                                 const int*   __restrict__ scale_p,
                                 float* __restrict__ Out,
                                 float* __restrict__ Attn)
{
    extern __shared__ float lds[];
    float* sc    = lds;                           // [16][SSTRIDE] scores -> probs
    float* red   = lds + SCORE_F;                 // [32] reduction scratch
    float* vtile = lds + SCORE_F + RED_F;         // [2][32][VSTRIDE] f32 V chunks

    const int tid  = threadIdx.x;      // 0..127 (4 wave32)
    const int wave = tid >> 5;
    const int lane = tid & 31;
    const int m16  = lane & 15;        // row (A/C) or column (B) within 16
    const int kgrp = lane >> 4;        // lane-half selects K sub-range per ISA layout

    const int bh = blockIdx.y;         // b*H + h
    const int q0 = blockIdx.x * 16;    // query-row tile base

    const float inv_scale = 1.0f / (float)scale_p[0];

    const float* Qb = Q + ((size_t)bh * S_ + q0) * D_;
    const float* Kb = K + (size_t)bh * S_ * D_;
    const float* Vb = V + (size_t)bh * S_ * D_;

    // ------------------------------------------------------------------
    // Phase 1: scores = (Q/scale) K^T  via v_wmma_f32_16x16x32_f16
// A (16x32 f16) lane layout: lanes0-15 M=0..15 K0..15,
// lanes16-31 same M, K16..31 (contiguous order chosen here to match B; the ISA table order differs).
    // ------------------------------------------------------------------
    v16h a0, a1;   // Q d=[0,32) and d=[32,64)
    {
        const float* qrow = Qb + m16 * D_;
        #pragma unroll
        for (int h = 0; h < 8; ++h) {
            a0[h]     = (_Float16)(qrow[     kgrp * 16 + h] * inv_scale);
            a0[8 + h] = (_Float16)(qrow[ 8 + kgrp * 16 + h] * inv_scale);
            a1[h]     = (_Float16)(qrow[32 + kgrp * 16 + h] * inv_scale);
            a1[8 + h] = (_Float16)(qrow[40 + kgrp * 16 + h] * inv_scale);
        }
    }

    // Each wave owns 512 key columns = 32 tiles of 16.
    for (int t = 0; t < 32; ++t) {
        const int n0 = wave * 512 + t * 16;
        const int n  = n0 + m16;                    // this lane's key column
        // B (32x16 f16): lanes0-15 hold K rows 0..15, lanes16-31 rows 16..31;
        // contiguous 64-byte read from one K row per lane.
        const float* krow = Kb + (size_t)n * D_ + kgrp * 16;
        if (t + 1 < 32) __builtin_prefetch(krow + 16 * D_, 0, 0);  // global_prefetch_b8
        v16h b0, b1;
        #pragma unroll
        for (int h = 0; h < 16; ++h) {
            b0[h] = (_Float16)krow[h];        // d in [0,32)
            b1[h] = (_Float16)krow[32 + h];   // d in [32,64)
        }
        v8f c = {};
        c = __builtin_amdgcn_wmma_f32_16x16x32_f16(false, a0, false, b0, (short)0, c, false, false);
        c = __builtin_amdgcn_wmma_f32_16x16x32_f16(false, a1, false, b1, (short)0, c, false, false);
        // C layout: VGPR r -> M = r (lanes0-15) / 8+r (lanes16-31), N = lane&15
        #pragma unroll
        for (int r = 0; r < 8; ++r)
            sc[(r + kgrp * 8) * SSTRIDE + n] = c[r];
    }
    __syncthreads();

    // ------------------------------------------------------------------
    // Phase 2: row-wise softmax in LDS; stream normalized P to Attn with
    // non-temporal stores (one-shot 537 MB stream; keep L2 for K/V reuse).
    // ------------------------------------------------------------------
    for (int r = 0; r < 16; ++r) {
        float* row = sc + r * SSTRIDE;
        float mv = -__builtin_inff();
        for (int c = tid; c < S_; c += 128) mv = fmaxf(mv, row[c]);
        #pragma unroll
        for (int off = 16; off >= 1; off >>= 1) mv = fmaxf(mv, __shfl_xor(mv, off, 32));
        if (lane == 0) red[wave] = mv;
        __syncthreads();
        mv = fmaxf(fmaxf(red[0], red[1]), fmaxf(red[2], red[3]));

        float s = 0.0f;
        for (int c = tid; c < S_; c += 128) {
            float e = __expf(row[c] - mv);
            row[c] = e;
            s += e;
        }
        #pragma unroll
        for (int off = 16; off >= 1; off >>= 1) s += __shfl_xor(s, off, 32);
        if (lane == 0) red[8 + wave] = s;
        __syncthreads();
        s = red[8] + red[9] + red[10] + red[11];

        const float inv = 1.0f / s;
        float* arow = Attn + ((size_t)bh * S_ + q0 + r) * S_;
        for (int c = tid; c < S_; c += 128) {
            float p = row[c] * inv;
            row[c] = p;                                // keep P in LDS for phase 3
            __builtin_nontemporal_store(p, arow + c);  // stream attention matrix
        }
    }

    // ------------------------------------------------------------------
    // Phase 3: Out = P V  (16x2048 · 2048x64), one 16-col slab per wave.
    // V chunks DMA'd into LDS by the Tensor Data Mover, double-buffered:
    // wave 0 issues chunk kt+1, waits tensorcnt<=1 (chunk kt complete,
    // TDM is in-order per wave), then a barrier releases all waves.
    // ------------------------------------------------------------------
    v8f acc = {};
    const int dcol = wave * 16;
    const unsigned vlds0 = (unsigned)(size_t)(void*)vtile;  // LDS byte address

    if (wave == 0)
        tdm_load_v_chunk(vlds0, (unsigned long long)(size_t)Vb);

    for (int kt = 0; kt < 64; ++kt) {
        __syncthreads();   // all waves done reading the buffer TDM writes next
        if (wave == 0) {
            if (kt + 1 < 64) {
                tdm_load_v_chunk(vlds0 + (unsigned)(((kt + 1) & 1) * VBUF_F * 4),
                                 (unsigned long long)(size_t)(Vb + (size_t)(kt + 1) * 32 * D_));
                __builtin_amdgcn_s_wait_tensorcnt(1);   // chunk kt landed
            } else {
                __builtin_amdgcn_s_wait_tensorcnt(0);   // last chunk landed
            }
        }
        __syncthreads();   // buffer kt visible to all waves

        const float* vt = vtile + (kt & 1) * VBUF_F;
        v16h a, b;
        const float* prow = sc + m16 * SSTRIDE + kt * 32;   // padded stride: bank-clean
        #pragma unroll
        for (int h = 0; h < 8; ++h) {
            a[h]     = (_Float16)prow[     kgrp * 16 + h];
            a[8 + h] = (_Float16)prow[ 8 + kgrp * 16 + h];
        }
        // B (32x16): lanes0-15 K rows 0..15, lanes16-31 rows 16..31.
        // VSTRIDE=65 => bank = (k + n) % 64: conflict-free across the wave.
        #pragma unroll
        for (int h = 0; h < 16; ++h)
            b[h] = (_Float16)vt[(kgrp * 16 + h) * VSTRIDE + dcol + m16];

        acc = __builtin_amdgcn_wmma_f32_16x16x32_f16(false, a, false, b, (short)0, acc, false, false);
    }

    #pragma unroll
    for (int r = 0; r < 8; ++r)
        Out[((size_t)bh * S_ + q0 + r + kgrp * 8) * D_ + dcol + m16] = acc[r];
}

extern "C" void kernel_launch(void* const* d_in, const int* in_sizes, int n_in,
                              void* d_out, int out_size, void* d_ws, size_t ws_size,
                              hipStream_t stream)
{
    const float* Q     = (const float*)d_in[0];
    const float* K     = (const float*)d_in[1];
    const float* V     = (const float*)d_in[2];
    const int*   scale = (const int*)  d_in[3];

    float* Out  = (float*)d_out;                            // [B,H,S,D]
    float* Attn = Out + (size_t)B_ * H_ * S_ * D_;          // [B,H,S,S]

    // Dynamic LDS exceeds the 64 KB default cap; raise it (deterministic,
    // non-stream host call — safe under graph capture).
    hipFuncSetAttribute((const void*)sdpa_wmma_kernel,
                        hipFuncAttributeMaxDynamicSharedMemorySize, SMEM_BYTES);

    dim3 grid(S_ / 16, B_ * H_);   // 128 query tiles x 32 (b,h)
    sdpa_wmma_kernel<<<grid, 128, SMEM_BYTES, stream>>>(Q, K, V, scale, Out, Attn);
}
